// Attention_80633716015544
// MI455X (gfx1250) — matrix operand order repaired and checked by reading, not compiled, not run on hardware
//
#include <hip/hip_runtime.h>
#include <hip/hip_bf16.h>

// PVT spatial-reduction attention for MI455X (gfx1250), bf16 WMMA everywhere.
// B=8, H=W=128, N=16384, C=64, nh=1, hd=64, SR=8, M=256.
//
// v2: K-split conv GEMM (2048 waves instead of 128 -> latency hiding),
//     fused reduce+bias+LayerNorm, q-projection fused into attention,
//     V-transpose fused into the KV GEMM store.

typedef __bf16 bf16_t;
typedef __attribute__((ext_vector_type(16))) __bf16 v16bf;
typedef __attribute__((ext_vector_type(8)))  float  v8f;

#define DEVINL static __device__ __forceinline__

DEVINL v8f wmma_bf16(v16bf a, v16bf b, v8f c) {
  // D = A(16x32) * B(32x16) + C, f32 accumulate
  return __builtin_amdgcn_wmma_f32_16x16x32_bf16(false, a, false, b,
                                                 (short)0, c, false, false);
}

DEVINL v8f vzero8() {
  v8f z;
#pragma unroll
  for (int r = 0; r < 8; ++r) z[r] = 0.0f;
  return z;
}

// ---- fragment loaders (layouts per CDNA5 ISA 7.12.2) ----

// A fragment 16x32 from row-major bf16 matrix. lane<16: M=lane, K=0..7,16..23;
// lane>=16: M=lane-16, K=8..15,24..31.
DEVINL v16bf load_a_bf16(const bf16_t* A, int lda, int m0, int k0) {
  int lane = threadIdx.x & 31;
  int row = lane & 15, hi = lane >> 4;
  const bf16_t* p = A + (size_t)(m0 + row) * lda + k0 + hi * 8;
  v16bf a;
#pragma unroll
  for (int j = 0; j < 8; ++j) a[j] = p[j];
#pragma unroll
  for (int j = 0; j < 8; ++j) a[j + 8] = p[16 + j];
  return a;
}

// A fragment 16x32 from row-major f32 matrix (convert to bf16 in regs).
DEVINL v16bf load_a_f32(const float* A, int lda, int m0, int k0) {
  int lane = threadIdx.x & 31;
  int row = lane & 15, hi = lane >> 4;
  const float* p = A + (size_t)(m0 + row) * lda + k0 + hi * 8;
  v16bf a;
#pragma unroll
  for (int j = 0; j < 8; ++j) a[j] = (bf16_t)p[j];
#pragma unroll
  for (int j = 0; j < 8; ++j) a[j + 8] = (bf16_t)p[16 + j];
  return a;
}

// B fragment 32x16 from an N-major matrix BT where BT[n][k] = B[k][n].
// lane<16: N=lane, K=0..7,16..23; lane>=16: N=lane-16, K=8..15,24..31.
DEVINL v16bf load_bt_bf16(const bf16_t* BT, int ldb, int n0, int k0) {
  int lane = threadIdx.x & 31;
  int n = lane & 15, hi = lane >> 4;
  const bf16_t* p = BT + (size_t)(n0 + n) * ldb + k0 + hi * 8;
  v16bf b;
#pragma unroll
  for (int j = 0; j < 16; ++j) b[j] = p[(j & 7) + ((j >> 3) << 4)];
  return b;
}

// ---------------------------------------------------------------------------
// 1) weight prep: transpose + convert to bf16 (N-major for B-fragment loads)
// ---------------------------------------------------------------------------
__global__ void prep_weights_kernel(const float* __restrict__ Wq,
                                    const float* __restrict__ Wkv,
                                    const float* __restrict__ Wproj,
                                    const float* __restrict__ srk,
                                    bf16_t* __restrict__ WqT,
                                    bf16_t* __restrict__ WkvT,
                                    bf16_t* __restrict__ WprojT,
                                    bf16_t* __restrict__ srkT) {
  int t = blockIdx.x * blockDim.x + threadIdx.x;
  if (t < 64 * 4096) {                 // srkT[co][kcol], srk is [kcol][co]
    int co = t >> 12, kcol = t & 4095;
    srkT[t] = (bf16_t)srk[(size_t)kcol * 64 + co];
  }
  if (t < 4096) {                      // WqT[n][k], WprojT[n][k]
    int n = t >> 6, k = t & 63;
    WqT[t]    = (bf16_t)Wq[k * 64 + n];
    WprojT[t] = (bf16_t)Wproj[k * 64 + n];
  }
  if (t < 8192) {                      // WkvT[n][k], n in [0,128)
    int n = t >> 6, k = t & 63;
    WkvT[t] = (bf16_t)Wkv[k * 128 + n];
  }
}

// ---------------------------------------------------------------------------
// 2) SR conv as K-split implicit-im2col GEMM: [2048 x 4096] * [4096 x 64].
//    16 K-chunks of 256 -> 2048 waves (latency hiding). f32 partials to ws.
//    Faithful to the reference's transpose(0,2,1).reshape(B,H,W,C) scramble.
// ---------------------------------------------------------------------------
__global__ void __launch_bounds__(256)
conv_partial_kernel(const float* __restrict__ x, const bf16_t* __restrict__ srkT,
                    float* __restrict__ convp) {
  int wave = (blockIdx.x * blockDim.x + threadIdx.x) >> 5;   // 0..2047
  int chunk = wave & 15;                // K chunk: [chunk*256, chunk*256+256)
  int m0 = (wave >> 4) * 16;            // row tile
  int lane = threadIdx.x & 31;
  int row = lane & 15, hi = lane >> 4;
  int gr = m0 + row;                    // b*256 + oh*16 + ow
  int b = gr >> 8, oh = (gr >> 4) & 15, ow = gr & 15;
  const float* xb = x + (size_t)b * (16384 * 64);

  v8f acc[4];
#pragma unroll
  for (int i = 0; i < 4; ++i) acc[i] = vzero8();

  int kbase = chunk * 256;
#pragma unroll 2
  for (int kk = 0; kk < 256; kk += 32) {
    int k0 = kbase + kk;
    v16bf a;
#pragma unroll
    for (int j = 0; j < 16; ++j) {
      int K = (j & 7) + ((j >> 3) << 4) + (hi << 3);
      int kcol = k0 + K;                // (kh*8+kw)*64 + ci
      int ci = kcol & 63, kw = (kcol >> 6) & 7, kh = kcol >> 9;
      int idx = (oh * 8 + kh) * 8192 + (ow * 8 + kw) * 64 + ci;
      // x_[b,h,w,c] = x[b, idx & 16383, idx >> 14]  (TF transpose scramble)
      a[j] = (bf16_t)xb[(size_t)(idx & 16383) * 64 + (idx >> 14)];
    }
#pragma unroll
    for (int i = 0; i < 4; ++i)
      acc[i] = wmma_bf16(a, load_bt_bf16(srkT, 4096, i * 16, k0), acc[i]);
  }
  int col = lane & 15;
  float* cp = convp + (size_t)chunk * (2048 * 64);
#pragma unroll
  for (int i = 0; i < 4; ++i)
#pragma unroll
    for (int r = 0; r < 8; ++r)
      cp[(size_t)(m0 + r + hi * 8) * 64 + i * 16 + col] = acc[i][r];
}

// ---------------------------------------------------------------------------
// 3) reduce partials + bias + LayerNorm (eps=1e-3). One wave per row.
// ---------------------------------------------------------------------------
__global__ void __launch_bounds__(256)
reduce_ln_kernel(const float* __restrict__ convp, const float* __restrict__ srb,
                 const float* __restrict__ gamma, const float* __restrict__ beta,
                 bf16_t* __restrict__ xln) {
  int rowi = (blockIdx.x * blockDim.x + threadIdx.x) >> 5;   // 0..2047
  int lane = threadIdx.x & 31;
  float v0 = srb[lane], v1 = srb[lane + 32];
#pragma unroll
  for (int ch = 0; ch < 16; ++ch) {
    const float* rp = convp + (size_t)ch * (2048 * 64) + (size_t)rowi * 64;
    v0 += rp[lane];
    v1 += rp[lane + 32];
  }
  float s = v0 + v1;
#pragma unroll
  for (int off = 1; off < 32; off <<= 1) s += __shfl_xor(s, off, 32);
  float mu = s * (1.0f / 64.0f);
  float d0 = v0 - mu, d1 = v1 - mu;
  float q = d0 * d0 + d1 * d1;
#pragma unroll
  for (int off = 1; off < 32; off <<= 1) q += __shfl_xor(q, off, 32);
  float rstd = rsqrtf(q * (1.0f / 64.0f) + 1e-3f);
  bf16_t* op = xln + (size_t)rowi * 64;
  op[lane]      = (bf16_t)(d0 * rstd * gamma[lane] + beta[lane]);
  op[lane + 32] = (bf16_t)(d1 * rstd * gamma[lane + 32] + beta[lane + 32]);
}

// ---------------------------------------------------------------------------
// 4) kv = x_ln @ Wkv : [2048 x 64] * [64 x 128].
//    k half -> kb[m][d] row-major; v half stored transposed -> vt[b][d][m].
// ---------------------------------------------------------------------------
__global__ void __launch_bounds__(256)
kv_gemm_kernel(const bf16_t* __restrict__ xln, const bf16_t* __restrict__ WkvT,
               bf16_t* __restrict__ kb, bf16_t* __restrict__ vt) {
  int wave = (blockIdx.x * blockDim.x + threadIdx.x) >> 5;   // 0..127
  int m0 = wave * 16;
  v8f acc[8];
#pragma unroll
  for (int i = 0; i < 8; ++i) acc[i] = vzero8();
#pragma unroll
  for (int s = 0; s < 2; ++s) {
    v16bf a = load_a_bf16(xln, 64, m0, s * 32);
#pragma unroll
    for (int i = 0; i < 8; ++i)
      acc[i] = wmma_bf16(a, load_bt_bf16(WkvT, 64, i * 16, s * 32), acc[i]);
  }
  int lane = threadIdx.x & 31;
  int col = lane & 15, hi = lane >> 4;
#pragma unroll
  for (int i = 0; i < 4; ++i) {                         // k half
#pragma unroll
    for (int r = 0; r < 8; ++r)
      kb[(size_t)(m0 + r + hi * 8) * 64 + i * 16 + col] = (bf16_t)acc[i][r];
  }
#pragma unroll
  for (int i = 4; i < 8; ++i) {                         // v half, transposed
    int d = (i - 4) * 16 + col;
#pragma unroll
    for (int r = 0; r < 8; ++r) {
      int gr = m0 + r + hi * 8;                         // b*256 + m
      int b = gr >> 8, m = gr & 255;
      vt[((size_t)b * 64 + d) * 256 + m] = (bf16_t)acc[i][r];
    }
  }
}

// ---------------------------------------------------------------------------
// 5) Fused q-projection + flash attention (M=256) + output projection + bias.
//    Wave handles 16 query rows; online softmax; q/P/O relaid through LDS.
// ---------------------------------------------------------------------------
__global__ void __launch_bounds__(256)
attn_kernel(const float* __restrict__ x, const bf16_t* __restrict__ WqT,
            const bf16_t* __restrict__ kb, const bf16_t* __restrict__ vt,
            const bf16_t* __restrict__ WprojT, const float* __restrict__ bproj,
            float* __restrict__ out) {
  __shared__ bf16_t scr[8][16 * 64];            // 2KB per wave, 16KB total
  int b = blockIdx.x >> 7;
  int qblk = blockIdx.x & 127;
  int w = threadIdx.x >> 5;
  int lane = threadIdx.x & 31;
  int col = lane & 15, hi = lane >> 4;
  int qrow0 = b * 16384 + qblk * 128 + w * 16;

  const bf16_t* Kb = kb + (size_t)b * 256 * 64;      // K[m][d]
  const bf16_t* Vb = vt + (size_t)b * 64 * 256;      // Vt[d][m]

  // ---- q = x @ Wq for this wave's 16 rows (C-layout -> LDS -> A-layout) ----
  {
    v8f qacc[4];
#pragma unroll
    for (int i = 0; i < 4; ++i) qacc[i] = vzero8();
#pragma unroll
    for (int s = 0; s < 2; ++s) {
      v16bf a = load_a_f32(x, 64, qrow0, s * 32);
#pragma unroll
      for (int i = 0; i < 4; ++i)
        qacc[i] = wmma_bf16(a, load_bt_bf16(WqT, 64, i * 16, s * 32), qacc[i]);
    }
#pragma unroll
    for (int i = 0; i < 4; ++i)
#pragma unroll
      for (int r = 0; r < 8; ++r)
        scr[w][(r + hi * 8) * 64 + i * 16 + col] = (bf16_t)qacc[i][r];
  }
  v16bf qa0 = load_a_bf16(&scr[w][0], 64, 0, 0);
  v16bf qa1 = load_a_bf16(&scr[w][0], 64, 0, 32);

  v8f m_run, l_run, o[4];
#pragma unroll
  for (int r = 0; r < 8; ++r) { m_run[r] = -1e30f; l_run[r] = 0.0f; }
#pragma unroll
  for (int i = 0; i < 4; ++i) o[i] = vzero8();

  bf16_t* ps = &scr[w][0];

#pragma unroll 1
  for (int t2 = 0; t2 < 8; ++t2) {               // pairs of 16-key tiles
#pragma unroll
    for (int sub = 0; sub < 2; ++sub) {
      int m0 = (t2 * 2 + sub) * 16;
      v8f s = vzero8();
      s = wmma_bf16(qa0, load_bt_bf16(Kb, 64, m0, 0), s);
      s = wmma_bf16(qa1, load_bt_bf16(Kb, 64, m0, 32), s);
      v8f rmax;
#pragma unroll
      for (int r = 0; r < 8; ++r) { s[r] *= 0.125f; rmax[r] = s[r]; }
      // row max across the 16 lanes of each half-wave
#pragma unroll
      for (int off = 1; off < 16; off <<= 1)
#pragma unroll
        for (int r = 0; r < 8; ++r)
          rmax[r] = fmaxf(rmax[r], __shfl_xor(rmax[r], off, 32));
      v8f p, rsum;
#pragma unroll
      for (int r = 0; r < 8; ++r) {
        float mnew = fmaxf(m_run[r], rmax[r]);
        float alpha = __expf(m_run[r] - mnew);
        p[r] = __expf(s[r] - mnew);
        rsum[r] = p[r];
        l_run[r] *= alpha;
        m_run[r] = mnew;
        rmax[r] = alpha;                          // reuse as alpha
      }
#pragma unroll
      for (int off = 1; off < 16; off <<= 1)
#pragma unroll
        for (int r = 0; r < 8; ++r)
          rsum[r] += __shfl_xor(rsum[r], off, 32);
#pragma unroll
      for (int r = 0; r < 8; ++r) l_run[r] += rsum[r];
#pragma unroll
      for (int i = 0; i < 4; ++i)
#pragma unroll
        for (int r = 0; r < 8; ++r) o[i][r] *= rmax[r];
      // stash P tile (C-layout -> row-major 16x32 region)
#pragma unroll
      for (int r = 0; r < 8; ++r)
        ps[(r + hi * 8) * 32 + sub * 16 + col] = (bf16_t)p[r];
    }
    // P (16x32) as A-fragment, V tiles as B-fragments
    v16bf pa = load_a_bf16(ps, 32, 0, 0);
#pragma unroll
    for (int i = 0; i < 4; ++i)
      o[i] = wmma_bf16(pa, load_bt_bf16(Vb, 256, i * 16, t2 * 32), o[i]);
  }

  // normalize and stash O (16x64) for the projection GEMM
#pragma unroll
  for (int i = 0; i < 4; ++i)
#pragma unroll
    for (int r = 0; r < 8; ++r)
      scr[w][(r + hi * 8) * 64 + i * 16 + col] = (bf16_t)(o[i][r] / l_run[r]);

  v8f acc[4];
#pragma unroll
  for (int i = 0; i < 4; ++i) acc[i] = vzero8();
#pragma unroll
  for (int s2 = 0; s2 < 2; ++s2) {
    v16bf oa = load_a_bf16(&scr[w][0], 64, 0, s2 * 32);
#pragma unroll
    for (int i = 0; i < 4; ++i)
      acc[i] = wmma_bf16(oa, load_bt_bf16(WprojT, 64, i * 16, s2 * 32), acc[i]);
  }
  float* op = out + (size_t)qrow0 * 64;
#pragma unroll
  for (int i = 0; i < 4; ++i) {
    float bia = bproj[i * 16 + col];
#pragma unroll
    for (int r = 0; r < 8; ++r)
      op[(size_t)(r + hi * 8) * 64 + i * 16 + col] = acc[i][r] + bia;
  }
}

// ---------------------------------------------------------------------------
extern "C" void kernel_launch(void* const* d_in, const int* in_sizes, int n_in,
                              void* d_out, int out_size, void* d_ws, size_t ws_size,
                              hipStream_t stream) {
  (void)in_sizes; (void)n_in; (void)out_size; (void)ws_size;
  const float* x     = (const float*)d_in[0];
  // d_in[1], d_in[2] are H=128, W=128 (known constants)
  const float* Wq    = (const float*)d_in[3];
  const float* Wkv   = (const float*)d_in[4];
  const float* srk   = (const float*)d_in[5];
  const float* srb   = (const float*)d_in[6];
  const float* gamma = (const float*)d_in[7];
  const float* beta  = (const float*)d_in[8];
  const float* Wproj = (const float*)d_in[9];
  const float* bproj = (const float*)d_in[10];
  float* out = (float*)d_out;

  char* ws = (char*)d_ws;
  size_t off = 0;
  auto alloc = [&](size_t bytes) -> void* {
    void* p = ws + off;
    off += (bytes + 255) & ~(size_t)255;
    return p;
  };
  bf16_t* WqT    = (bf16_t*)alloc(4096 * 2);
  bf16_t* WkvT   = (bf16_t*)alloc(8192 * 2);
  bf16_t* WprojT = (bf16_t*)alloc(4096 * 2);
  bf16_t* srkT   = (bf16_t*)alloc((size_t)262144 * 2);
  float*  convp  = (float*)alloc((size_t)16 * 2048 * 64 * 4);   // 8 MB
  bf16_t* xln    = (bf16_t*)alloc((size_t)2048 * 64 * 2);
  bf16_t* kbuf   = (bf16_t*)alloc((size_t)2048 * 64 * 2);
  bf16_t* vtb    = (bf16_t*)alloc((size_t)8 * 64 * 256 * 2);

  prep_weights_kernel<<<1024, 256, 0, stream>>>(Wq, Wkv, Wproj, srk,
                                                WqT, WkvT, WprojT, srkT);
  conv_partial_kernel<<<256, 256, 0, stream>>>(x, srkT, convp);
  reduce_ln_kernel<<<256, 256, 0, stream>>>(convp, srb, gamma, beta, xln);
  kv_gemm_kernel<<<16, 256, 0, stream>>>(xln, WkvT, kbuf, vtb);
  attn_kernel<<<1024, 256, 0, stream>>>(x, WqT, kbuf, vtb, WprojT, bproj, out);
}
